// TabPDLHead_15418932593141
// MI455X (gfx1250) — matrix operand order repaired and checked by reading, not compiled, not run on hardware
//
#include <hip/hip_runtime.h>
#include <hip/hip_bf16.h>

// Problem constants (match reference)
#define B_ 4
#define N_ 4096
#define M_ 2048
#define D_ 512
#define C_ 10
#define TOPK_ 64

typedef __attribute__((ext_vector_type(16))) __bf16 v16bf;
typedef __attribute__((ext_vector_type(8)))  float  v8f;
typedef __attribute__((ext_vector_type(4)))  unsigned int v4u;
typedef __attribute__((ext_vector_type(8)))  int    v8i_;
typedef __attribute__((ext_vector_type(4)))  int    v4i_;

union BF16Frag {
    v16bf v;
    uint4 q[2];
};

// ---------------------------------------------------------------------------
// K0a: f32 -> bf16 conversion for weight matrices
// ---------------------------------------------------------------------------
__global__ void cvt_bf16_kernel(const float* __restrict__ src,
                                __bf16* __restrict__ dst, int n) {
    int i = blockIdx.x * blockDim.x + threadIdx.x;
    if (i < n) dst[i] = (__bf16)src[i];
}

// ---------------------------------------------------------------------------
// K0b: per-batch class prior + 1/max(1-prior,1e-6)
// ---------------------------------------------------------------------------
__global__ void prior_kernel(const int* __restrict__ y,
                             const unsigned char* __restrict__ mask,
                             float* __restrict__ prior,
                             float* __restrict__ invd) {
    __shared__ float cnts[C_];
    int b = blockIdx.x;
    int tid = threadIdx.x;
    if (tid < C_) cnts[tid] = 0.0f;
    __syncthreads();
    for (int i = tid; i < N_; i += blockDim.x) {
        if (mask[b * N_ + i]) atomicAdd(&cnts[y[b * N_ + i]], 1.0f);
    }
    __syncthreads();
    if (tid < C_) {
        float tot = 0.0f;
        #pragma unroll
        for (int c = 0; c < C_; ++c) tot += cnts[c];
        float p = cnts[tid] / fmaxf(tot, 1.0f);
        prior[b * C_ + tid] = p;
        invd[b * C_ + tid] = 1.0f / fmaxf(1.0f - p, 1e-6f);
    }
}

// ---------------------------------------------------------------------------
// K1: fused LayerNorm + projection  Out[row,e] = sum_d LN(H[row,d]) * W[e,d]
// Block: 256 threads = 8 waves, tile = 32 rows x 512 cols.
// ---------------------------------------------------------------------------
__global__ __launch_bounds__(256, 2)
void ln_project_kernel(const float* __restrict__ H,
                       const float* __restrict__ lnw,
                       const float* __restrict__ lnb,
                       const __bf16* __restrict__ Wb,
                       __bf16* __restrict__ Out) {
    __shared__ __bf16 As[32][528];   // padded stride (1056B rows, 16B aligned)

    const int row0 = blockIdx.x * 32;
    const int tid = threadIdx.x;

    // ---- Phase A: LayerNorm 32 rows, 8 lanes per row ----
    {
        const int rl  = tid >> 3;   // 0..31 row in tile
        const int sub = tid & 7;    // 0..7 column group
        const float* hrow = H + (size_t)(row0 + rl) * D_;
        float s = 0.0f, s2 = 0.0f;
        #pragma unroll 4
        for (int j = 0; j < 64; ++j) {
            float x = hrow[sub * 64 + j];
            s += x; s2 += x * x;
        }
        #pragma unroll
        for (int d = 1; d < 8; d <<= 1) {
            s  += __shfl_xor(s,  d, 32);
            s2 += __shfl_xor(s2, d, 32);
        }
        const float mean = s * (1.0f / (float)D_);
        const float var  = s2 * (1.0f / (float)D_) - mean * mean;
        const float rs   = rsqrtf(var + 1e-5f);
        #pragma unroll 4
        for (int j = 0; j < 64; ++j) {
            int col = sub * 64 + j;
            float x  = hrow[col];
            float xn = (x - mean) * rs * lnw[col] + lnb[col];
            As[rl][col] = (__bf16)xn;
        }
    }
    __syncthreads();

    // ---- Phase B: WMMA GEMM ----
    const int wave = tid >> 5;
    const int lane = tid & 31;
    const int rt   = wave >> 2;          // row tile 0/1
    const int cq   = wave & 3;           // 128-col quad
    const int mBase = rt * 16;
    const int l16  = lane & 15;
    const bool hiL = lane >= 16;

    for (int t = 0; t < 8; ++t) {
        const int col0 = cq * 128 + t * 16;
        v8f acc = {};
        #pragma unroll 4
        for (int ks = 0; ks < D_; ks += 32) {
            BF16Frag a, bfr;
            const __bf16* ap = &As[mBase + l16][ks + (hiL ? 16 : 0)];
            a.q[0] = *(const uint4*)ap;
            a.q[1] = *(const uint4*)(ap + 8);
            const __bf16* bp = Wb + (size_t)(col0 + l16) * D_ + ks + (hiL ? 16 : 0);
            bfr.q[0] = *(const uint4*)bp;
            bfr.q[1] = *(const uint4*)(bp + 8);
            acc = __builtin_amdgcn_wmma_f32_16x16x32_bf16(
                false, a.v, false, bfr.v, (short)0, acc, false, false);
        }
        const int n = col0 + l16;
        #pragma unroll
        for (int r = 0; r < 8; ++r) {
            int m = mBase + r + (hiL ? 8 : 0);
            Out[(size_t)(row0 + m) * D_ + n] = (__bf16)acc[r];
        }
    }
}

// ---------------------------------------------------------------------------
// K2: fused QK^T + sigmoid + exact top-64 threshold + class aggregation.
// Grid: B * (M/16) blocks. Block: 512 threads = 16 waves.
// Q strip (16x512 bf16) is staged into padded LDS by the Tensor Data Mover
// (tensor_load_to_lds with D# pad feature reproducing the 528-elem stride).
// ---------------------------------------------------------------------------
__global__ __launch_bounds__(512, 1)
void fused_head_kernel(const __bf16* __restrict__ Qb,
                       const __bf16* __restrict__ Kb,
                       const int* __restrict__ y,
                       const unsigned char* __restrict__ mask,
                       const float* __restrict__ prior,
                       const float* __restrict__ invd,
                       const float* __restrict__ tauP,
                       const float* __restrict__ biasP,
                       float* __restrict__ out) {
    __shared__ float  gamma[16][N_ + 4];   // pad stride 4100 (not mult of 64)
    __shared__ __bf16 Qs[16][528];         // 1056B row stride (512 data + 32B pad)

    const int b    = blockIdx.x >> 7;        // / (M/16)
    const int mt   = blockIdx.x & 127;
    const int row0 = mt * 16;                // query row base within batch
    const int tid  = threadIdx.x;
    const int wave = tid >> 5;
    const int lane = tid & 31;
    const int l16  = lane & 15;
    const bool hiL = lane >= 16;

    // scalar params
    const float tp   = *tauP;
    const float tau  = (tp > 20.0f ? tp : log1pf(__expf(tp))) + 1e-6f;
    const float bias = *biasP;

    // ---- TDM: stage 16x512 bf16 Q strip into LDS with row padding ----
    if (wave == 0) {
        const unsigned lds_off = (unsigned)(reinterpret_cast<uintptr_t>(&Qs[0][0]));
        const unsigned long long ga =
            (unsigned long long)reinterpret_cast<uintptr_t>(
                Qb + (size_t)(b * M_ + row0) * D_);
        // D# group 0: count=1 | lds_addr | global_addr[56:0] | type=2
        v4u g0 = { 1u,
                   lds_off,
                   (unsigned)(ga & 0xFFFFFFFFu),
                   (unsigned)((ga >> 32) & 0x01FFFFFFu) | 0x80000000u };
        // D# group 1:
        //  w0: data_size=1(2B)<<16 | pad_enable<<20 | pad_interval=7(256dw)<<22
        //      | pad_amount=7(8dw)<<25   -> pad 32B after each 1024B row
        //  w1: atomic_barrier_addr=0 | tensor_dim0(lo16)<<16
        //  w2: tensor_dim0(hi16) | tensor_dim1(lo16)<<16
        //  w3: tensor_dim1(hi16) | tile_dim0<<16
        //  w4: tile_dim1=16 | tile_dim2=0
        //  w5: tensor_dim0_stride(lo32)   w6,w7: stride hi / dim1_stride = 0
        v8i_ g1 = { (int)0x0FD10000,
                    (int)((D_ & 0xFFFF) << 16),
                    (int)(((unsigned)(B_ * M_) & 0xFFFFu) << 16),
                    (int)((D_ & 0xFFFF) << 16),
                    16,
                    (int)D_,
                    0, 0 };
        v4i_ gz  = { 0, 0, 0, 0 };
#if defined(__clang_major__) && (__clang_major__ >= 23)
        v8i_ gz8 = { 0, 0, 0, 0, 0, 0, 0, 0 };
        __builtin_amdgcn_tensor_load_to_lds(g0, g1, gz, gz, gz8, 0);
#else
        __builtin_amdgcn_tensor_load_to_lds(g0, g1, gz, gz, 0);
#endif
        __builtin_amdgcn_s_wait_tensorcnt(0);
    }
    __syncthreads();

    // ---- GEMM phase: wave owns cols [wave*256, wave*256+256) ----
    const __bf16* Kbase = Kb + (size_t)b * N_ * D_;
    const unsigned char* mrow = mask + b * N_;
    for (int t = 0; t < 16; ++t) {
        const int n0 = wave * 256 + t * 16;
        if (t + 1 < 16) {
            // warm caches for next K tile (global_prefetch_b8)
            __builtin_prefetch(Kbase + (size_t)(n0 + 16 + l16) * D_, 0, 1);
        }
        v8f acc = {};
        const __bf16* krow = Kbase + (size_t)(n0 + l16) * D_ + (hiL ? 16 : 0);
        #pragma unroll 4
        for (int ks = 0; ks < D_; ks += 32) {
            BF16Frag a, bfr;
            const __bf16* ap = &Qs[l16][ks + (hiL ? 16 : 0)];
            a.q[0] = *(const uint4*)ap;
            a.q[1] = *(const uint4*)(ap + 8);
            const __bf16* bp = krow + ks;
            bfr.q[0] = *(const uint4*)bp;
            bfr.q[1] = *(const uint4*)(bp + 8);
            acc = __builtin_amdgcn_wmma_f32_16x16x32_bf16(
                false, a.v, false, bfr.v, (short)0, acc, false, false);
        }
        const int n = n0 + l16;
        const bool mk = mrow[n] != 0;
        #pragma unroll
        for (int r = 0; r < 8; ++r) {
            int m = r + (hiL ? 8 : 0);
            float logit = tau * acc[r] + bias;
            float g = mk ? (1.0f / (1.0f + __expf(-logit))) : 0.0f;
            gamma[m][n] = g;
        }
    }
    __syncthreads();

    // ---- Selection + aggregation: wave w handles query row w ----
    const float* grow = gamma[wave];
    const float4* g4  = (const float4*)grow;   // row stride 16400B = 1025*16B

    // exact 64th-largest via binary search over ordered uint bits (g >= 0)
    unsigned lo = 0u, hiB = 0x3F800000u;   // gamma in [0,1]
    while (lo < hiB) {
        unsigned mid = lo + ((hiB - lo + 1u) >> 1);
        int c = 0;
        for (int j = lane; j < N_ / 4; j += 32) {
            float4 gv = g4[j];
            c += (__float_as_uint(gv.x) >= mid) ? 1 : 0;
            c += (__float_as_uint(gv.y) >= mid) ? 1 : 0;
            c += (__float_as_uint(gv.z) >= mid) ? 1 : 0;
            c += (__float_as_uint(gv.w) >= mid) ? 1 : 0;
        }
        #pragma unroll
        for (int d = 16; d; d >>= 1) c += __shfl_xor(c, d, 32);
        if (c >= TOPK_) lo = mid; else hiB = mid - 1u;
    }
    const float kth = __uint_as_float(lo);

    // per-batch class constants
    float pr[C_], ivd[C_];
    #pragma unroll
    for (int ci = 0; ci < C_; ++ci) {
        pr[ci]  = prior[b * C_ + ci];
        ivd[ci] = invd[b * C_ + ci];
    }

    float pos[C_], Ac[C_];
    #pragma unroll
    for (int ci = 0; ci < C_; ++ci) { pos[ci] = 0.0f; Ac[ci] = 0.0f; }
    float T = 0.0f, cnt = 0.0f;

    const int* yrow = y + b * N_;
    const int4* y4  = (const int4*)yrow;
    for (int j = lane; j < N_ / 4; j += 32) {
        float4 gv = g4[j];
        int4   yv = y4[j];
        unsigned mv = *(const unsigned int*)(mrow + 4 * j);
        float ge[4] = { gv.x, gv.y, gv.z, gv.w };
        int   ye[4] = { yv.x, yv.y, yv.z, yv.w };
        #pragma unroll
        for (int e = 0; e < 4; ++e) {
            float g = ge[e];
            if (g >= kth && ((mv >> (8 * e)) & 0xFFu)) {
                int   yc = ye[e];
                float w0 = 1.0f - g;
                float iv = 0.0f;
                #pragma unroll
                for (int ci = 0; ci < C_; ++ci) iv = (ci == yc) ? ivd[ci] : iv;
                float wA = w0 * iv;
                T   += wA;
                cnt += 1.0f;
                #pragma unroll
                for (int ci = 0; ci < C_; ++ci) {
                    if (ci == yc) { pos[ci] += g; Ac[ci] += wA; }
                }
            }
        }
    }

    // wave reduction
    #pragma unroll
    for (int ci = 0; ci < C_; ++ci) {
        #pragma unroll
        for (int d = 16; d; d >>= 1) {
            pos[ci] += __shfl_xor(pos[ci], d, 32);
            Ac[ci]  += __shfl_xor(Ac[ci],  d, 32);
        }
    }
    #pragma unroll
    for (int d = 16; d; d >>= 1) {
        T   += __shfl_xor(T,   d, 32);
        cnt += __shfl_xor(cnt, d, 32);
    }

    if (lane == 0) {
        float dq = fmaxf(cnt, 1.0f);
        float p[C_];
        float sum = 0.0f;
        #pragma unroll
        for (int ci = 0; ci < C_; ++ci) {
            float v = (pos[ci] + pr[ci] * (T - Ac[ci])) / dq;
            v = fmaxf(v, 0.0f);
            p[ci] = v;
            sum += v;
        }
        float inv = 1.0f / fmaxf(sum, 1e-12f);
        size_t qrow = (size_t)(b * M_ + row0 + wave);
        #pragma unroll
        for (int ci = 0; ci < C_; ++ci) out[qrow * C_ + ci] = p[ci] * inv;
    }
}

// ---------------------------------------------------------------------------
// Launch
// ---------------------------------------------------------------------------
extern "C" void kernel_launch(void* const* d_in, const int* in_sizes, int n_in,
                              void* d_out, int out_size, void* d_ws, size_t ws_size,
                              hipStream_t stream) {
    (void)in_sizes; (void)n_in; (void)out_size; (void)ws_size;

    const float*         H_support = (const float*)d_in[0];          // [B,N,D]
    const float*         H_query   = (const float*)d_in[1];          // [B,M,D]
    const int*           y_support = (const int*)d_in[2];            // [B,N]
    const unsigned char* smask     = (const unsigned char*)d_in[3];  // [B,N] bool
    const float*         ln_w      = (const float*)d_in[4];
    const float*         ln_b      = (const float*)d_in[5];
    const float*         WQ        = (const float*)d_in[6];          // [D,D]
    const float*         WK        = (const float*)d_in[7];
    const float*         tau_param = (const float*)d_in[8];
    const float*         bias      = (const float*)d_in[9];
    float*               out       = (float*)d_out;                  // [B,M,C]

    // workspace layout (all 256B aligned)
    char* ws = (char*)d_ws;
    __bf16* Wqb  = (__bf16*)(ws);                         // 512KB
    __bf16* Wkb  = (__bf16*)(ws + (size_t)524288);        // 512KB
    __bf16* Qb   = (__bf16*)(ws + (size_t)1048576);       // 8MB  [B*M, D]
    __bf16* Kb   = (__bf16*)(ws + (size_t)9437184);       // 16MB [B*N, D]
    float*  pri  = (float*)(ws + (size_t)26214400);       // [B,C]
    float*  ivd  = (float*)(ws + (size_t)26214656);       // [B,C]

    // weights to bf16
    cvt_bf16_kernel<<<(D_ * D_ + 255) / 256, 256, 0, stream>>>(WQ, Wqb, D_ * D_);
    cvt_bf16_kernel<<<(D_ * D_ + 255) / 256, 256, 0, stream>>>(WK, Wkb, D_ * D_);

    // per-batch priors
    prior_kernel<<<B_, 256, 0, stream>>>(y_support, smask, pri, ivd);

    // LN + projections (Q then K)
    ln_project_kernel<<<(B_ * M_) / 32, 256, 0, stream>>>(H_query,   ln_w, ln_b, Wqb, Qb);
    ln_project_kernel<<<(B_ * N_) / 32, 256, 0, stream>>>(H_support, ln_w, ln_b, Wkb, Kb);

    // fused logits + sigmoid + top-k + aggregation
    fused_head_kernel<<<B_ * (M_ / 16), 512, 0, stream>>>(
        Qb, Kb, y_support, smask, pri, ivd, tau_param, bias, out);
}
